// MessagePassing_27169963114586
// MI455X (gfx1250) — matrix operand order repaired and checked by reading, not compiled, not run on hardware
//
#include <hip/hip_runtime.h>

// ---------------- problem constants (from setup_inputs) ----------------
constexpr int BB = 4;      // batch
constexpr int MM = 4096;   // nodes m
constexpr int EE = 2048;   // hyperedges e
constexpr int DD = 256;    // feature dim d

typedef __bf16 bf16_t;
typedef bf16_t v4bf  __attribute__((ext_vector_type(4)));
typedef bf16_t v8bf  __attribute__((ext_vector_type(8)));
typedef bf16_t v16bf __attribute__((ext_vector_type(16)));
typedef float  v8f   __attribute__((ext_vector_type(8)));
typedef int    v4i   __attribute__((vector_size(16)));   // matches builtin param type

#if __has_builtin(__builtin_amdgcn_global_load_async_to_lds_b128) && \
    __has_builtin(__builtin_amdgcn_s_wait_asynccnt)
#define HAS_ASYNC 1
#else
#define HAS_ASYNC 0
#endif

// ---------------- workspace layout (bytes, all 256B aligned) ----------------
constexpr size_t O_INCB = 0;                                    // bf16 [b][m][e]
constexpr size_t O_FEATB = O_INCB + (size_t)BB*MM*EE*2;         // bf16 [b][m][d]
constexpr size_t O_FEATT = O_FEATB + (size_t)BB*MM*DD*2;        // bf16 [b][d][m]
constexpr size_t O_WATT  = O_FEATT + (size_t)BB*MM*DD*2;        // bf16 [d][d]
constexpr size_t O_VCW   = O_WATT + (size_t)DD*DD*2;
constexpr size_t O_ECW   = O_VCW  + (size_t)DD*DD*2;
constexpr size_t O_ST    = O_ECW  + (size_t)DD*DD*2;            // bf16 S^T [b][d][m]
constexpr size_t O_ES    = O_ST   + (size_t)BB*DD*MM*2;         // f32 [b][e][d]
constexpr size_t O_AG    = O_ES   + (size_t)BB*EE*DD*4;         // f32 [b][e][d]
constexpr size_t O_P     = O_AG   + (size_t)BB*EE*DD*4;         // bf16 [b][e][d]
constexpr size_t O_EFT   = O_P    + (size_t)BB*EE*DD*2;         // bf16 EF^T [b][d][e]
constexpr size_t O_WN    = O_EFT  + (size_t)BB*DD*EE*2;         // bf16 [b][m][d]
constexpr size_t O_PM    = O_WN   + (size_t)BB*MM*DD*2;         // f32 [b][16][d]
constexpr size_t O_PS    = O_PM   + (size_t)BB*16*DD*4;         // f32 [b][16][d]
constexpr size_t O_MX    = O_PS   + (size_t)BB*16*DD*4;         // f32 [b][d]
constexpr size_t O_ZX    = O_MX   + (size_t)BB*DD*4;            // f32 [b][d]

static __device__ inline v16bf pack16(v8bf a, v8bf b) {
  v16bf r;
#pragma unroll
  for (int i = 0; i < 8; ++i) { r[i] = a[i]; r[i + 8] = b[i]; }
  return r;
}

// 16-byte global -> LDS tile copy: async (gfx1250 GLOBAL_LOAD_ASYNC_TO_LDS_B128)
// when available, else sync through VGPRs.
static __device__ inline void gld16(bf16_t* lds, const bf16_t* g) {
#if HAS_ASYNC
  __builtin_amdgcn_global_load_async_to_lds_b128(
      (__attribute__((address_space(1))) v4i*)(g),
      (__attribute__((address_space(3))) v4i*)(lds), 0, 0);
#else
  *(v8bf*)lds = *(const v8bf*)g;
#endif
}

static __device__ inline void wait_async_tiles() {
#if HAS_ASYNC
  __builtin_amdgcn_s_wait_asynccnt(0);
#endif
}

// ---------------- conversion kernels ----------------
__global__ void k_conv_inc(const float4* __restrict__ in, float4* __restrict__ echo,
                           v4bf* __restrict__ outb, int n4) {
  int i = blockIdx.x * blockDim.x + threadIdx.x;
  if (i >= n4) return;
  float4 v = in[i];
  echo[i] = v;
  v4bf h; h[0] = (bf16_t)v.x; h[1] = (bf16_t)v.y; h[2] = (bf16_t)v.z; h[3] = (bf16_t)v.w;
  outb[i] = h;
}

__global__ void k_conv_w(const float* __restrict__ in, bf16_t* __restrict__ out, int n) {
  int i = blockIdx.x * blockDim.x + threadIdx.x;
  if (i < n) out[i] = (bf16_t)in[i];
}

// feat f32 [b][m][d] -> featB bf16 [b][m][d] and featT bf16 [b][d][m]
__global__ void k_feat(const float* __restrict__ feat, bf16_t* __restrict__ fb,
                       bf16_t* __restrict__ ft) {
  __shared__ float t[32][33];
  int bz = blockIdx.z;
  int c0 = blockIdx.x * 32, r0 = blockIdx.y * 32;
  int tx = threadIdx.x, ty = threadIdx.y;
  const float* F = feat + (size_t)bz * MM * DD;
  bf16_t* FB = fb + (size_t)bz * MM * DD;
  bf16_t* FT = ft + (size_t)bz * DD * MM;
#pragma unroll
  for (int i = 0; i < 4; ++i) {
    int r = ty + i * 8;
    float v = F[(size_t)(r0 + r) * DD + c0 + tx];
    t[r][tx] = v;
    FB[(size_t)(r0 + r) * DD + c0 + tx] = (bf16_t)v;
  }
  __syncthreads();
#pragma unroll
  for (int i = 0; i < 4; ++i) {
    int r = ty + i * 8;
    FT[(size_t)(c0 + r) * MM + r0 + tx] = (bf16_t)t[tx][r];
  }
}

// ---------------- softmax over e (per batch, per column d) ----------------
__global__ void k_smpart(const float* __restrict__ ES, float* __restrict__ pm,
                         float* __restrict__ ps) {
  int bz = blockIdx.y, c = blockIdx.x, tid = threadIdx.x;
  const float* e = ES + (size_t)bz * EE * DD + (size_t)c * 128 * DD;
  float m = -1e30f, s = 0.f;
  for (int r = 0; r < 128; ++r) {
    float v = e[(size_t)r * DD + tid];
    float nm = fmaxf(m, v);
    s = s * __expf(m - nm) + __expf(v - nm);
    m = nm;
  }
  pm[(size_t)(bz * 16 + c) * DD + tid] = m;
  ps[(size_t)(bz * 16 + c) * DD + tid] = s;
}

__global__ void k_smcomb(const float* __restrict__ pm, const float* __restrict__ ps,
                         float* __restrict__ Mx, float* __restrict__ Zx) {
  int bz = blockIdx.x, tid = threadIdx.x;
  float m = -1e30f;
  for (int c = 0; c < 16; ++c) m = fmaxf(m, pm[(size_t)(bz * 16 + c) * DD + tid]);
  float s = 0.f;
  for (int c = 0; c < 16; ++c)
    s += ps[(size_t)(bz * 16 + c) * DD + tid] * __expf(pm[(size_t)(bz * 16 + c) * DD + tid] - m);
  Mx[(size_t)bz * DD + tid] = m;
  Zx[(size_t)bz * DD + tid] = s;
}

// P = softmax(ES) * AG  -> bf16
__global__ void k_attnmul(const float* __restrict__ ES, const float* __restrict__ AG,
                          const float* __restrict__ Mx, const float* __restrict__ Zx,
                          bf16_t* __restrict__ P) {
  int bz = blockIdx.y;
  size_t base = (size_t)bz * EE * DD;
  size_t idx = (size_t)blockIdx.x * 256 + threadIdx.x;
  int col = (int)(idx & (DD - 1));
  float a = __expf(ES[base + idx] - Mx[(size_t)bz * DD + col]) / Zx[(size_t)bz * DD + col];
  P[base + idx] = (bf16_t)(a * AG[base + idx]);
}

// ---------------- WMMA GEMM: C[M,N] = A[M,K] * BT[N,K]^T ----------------
// block 128x128x(k-step 32), 8 waves (wave32), each wave 4x2 16x16 WMMA tiles.
// Double-buffered LDS; tiles streamed with GLOBAL_LOAD_ASYNC_TO_LDS_B128 when available.
template <bool TRANS_A, bool FUSE2>
__global__ __launch_bounds__(256) void gemm_k(
    const bf16_t* __restrict__ A, size_t sA, int lda,        // TRANS_A: A phys [K][M], lda = phys row len
    const bf16_t* __restrict__ B1, size_t sB1, int ldb1,     // BT1 [N][K]
    const bf16_t* __restrict__ B2, size_t sB2, int ldb2,     // BT2 [N][K] (FUSE2)
    float* __restrict__ C1f, size_t sC1f,                    // f32 [M][N] (or blend dst)
    bf16_t* __restrict__ C1t, size_t sC1t, int ldct,         // bf16 transposed [N][M]
    bf16_t* __restrict__ C1h, size_t sC1h,                   // bf16 [M][N]
    float* __restrict__ C2f, size_t sC2f,                    // f32 second output (FUSE2)
    const float* __restrict__ blendSrc, size_t sBlend,
    const float* __restrict__ alphaP, const int* __restrict__ nP,
    int K, int ldc) {
  constexpr int SA = 40, SB = 40, TSZ = 128 * 40;
  __shared__ bf16_t Al[2 * TSZ];
  __shared__ bf16_t Bl1[2 * TSZ];
  __shared__ bf16_t Bl2[FUSE2 ? 2 * TSZ : 8];

  const int tid = threadIdx.x;
  const int bz = blockIdx.z;
  const int m0 = blockIdx.x * 128;
  const int n0 = blockIdx.y * 128;
  const int wid = tid >> 5, lane = tid & 31;
  const int lm = lane & 15, lh = lane >> 4;
  const int wm = (wid >> 2) * 64, wn = (wid & 3) * 32;

  const bf16_t* Ab = A + sA * bz;
  const bf16_t* B1b = B1 + sB1 * bz;
  const bf16_t* B2b = FUSE2 ? (B2 + sB2 * bz) : nullptr;

  auto loadTiles = [&](int buf, int k0) {
    bf16_t* al = &Al[buf * TSZ];
    bf16_t* bl1 = &Bl1[buf * TSZ];
#pragma unroll
    for (int p = 0; p < 2; ++p) {
      int lin = (tid + p * 256) * 8;
      int r = lin >> 5, c = lin & 31;
      if constexpr (!TRANS_A) {
        gld16(al + r * SA + c, Ab + (size_t)(m0 + r) * lda + k0 + c);
      } else {
        int pr = lin >> 7, pc = lin & 127;  // phys row (K), phys col (M)
        v8bf v = *(const v8bf*)(Ab + (size_t)(k0 + pr) * lda + m0 + pc);
#pragma unroll
        for (int j = 0; j < 8; ++j) al[(pc + j) * SA + pr] = v[j];
      }
      gld16(bl1 + r * SB + c, B1b + (size_t)(n0 + r) * ldb1 + k0 + c);
      if constexpr (FUSE2) {
        bf16_t* bl2 = &Bl2[buf * TSZ];
        gld16(bl2 + r * SB + c, B2b + (size_t)(n0 + r) * ldb2 + k0 + c);
      }
    }
  };

  const v8f vzero = {0.f, 0.f, 0.f, 0.f, 0.f, 0.f, 0.f, 0.f};
  v8f acc1[4][2], acc2[4][2];
#pragma unroll
  for (int t = 0; t < 4; ++t)
#pragma unroll
    for (int u = 0; u < 2; ++u) { acc1[t][u] = vzero; acc2[t][u] = vzero; }

  const int nk = K >> 5;
  loadTiles(0, 0);
  for (int i = 0; i < nk; ++i) {
    const int cur = i & 1;
    wait_async_tiles();     // my async tile writes landed in LDS
    __syncthreads();        // everyone's landed; prev reads of other buffer done
    if (i + 1 < nk) loadTiles(cur ^ 1, (i + 1) << 5);  // stream next tile during compute

    const bf16_t* al = &Al[cur * TSZ];
    const bf16_t* bl1 = &Bl1[cur * TSZ];
    const bf16_t* bl2 = &Bl2[FUSE2 ? cur * TSZ : 0];

    // ---- fragments (documented CDNA5 wave32 layouts) ----
    v16bf af[4], bf1[2], bf2[2];
#pragma unroll
    for (int t = 0; t < 4; ++t) {
      int row = (wm + t * 16 + lm) * SA;
      v8bf x0 = *(const v8bf*)&al[row + lh * 8];        // K 0..7 / 8..15
      v8bf x1 = *(const v8bf*)&al[row + 16 + lh * 8];   // K 16..23 / 24..31
      af[t] = pack16(x0, x1);
    }
#pragma unroll
    for (int u = 0; u < 2; ++u) {
      int col = (wn + u * 16 + lm) * SB + lh * 8;       // K 0..7,16..23 / 8..15,24..31
      bf1[u] = pack16(*(const v8bf*)&bl1[col], *(const v8bf*)&bl1[col + 16]);
      if constexpr (FUSE2)
        bf2[u] = pack16(*(const v8bf*)&bl2[col], *(const v8bf*)&bl2[col + 16]);
    }
#pragma unroll
    for (int t = 0; t < 4; ++t)
#pragma unroll
      for (int u = 0; u < 2; ++u) {
        acc1[t][u] = __builtin_amdgcn_wmma_f32_16x16x32_bf16(
            false, af[t], false, bf1[u], (short)0, acc1[t][u], false, false);
        if constexpr (FUSE2)
          acc2[t][u] = __builtin_amdgcn_wmma_f32_16x16x32_bf16(
              false, af[t], false, bf2[u], (short)0, acc2[t][u], false, false);
      }
    // no trailing barrier: next iteration's wait+barrier protects buffer reuse
  }

  // ---- epilogue ----
  float ca = 0.f, cb = 0.f;
  if (blendSrc) {
    float a = *alphaP;
    int n = *nP;
    ca = powf(a, (float)n);
    cb = 1.f - ca;
  }
#pragma unroll
  for (int t = 0; t < 4; ++t)
#pragma unroll
    for (int u = 0; u < 2; ++u) {
      int mB = m0 + wm + t * 16 + lh * 8;     // 8 consecutive M rows per lane
      int nG = n0 + wn + u * 16 + lm;
      if (C1t) {
        v8bf h;
#pragma unroll
        for (int r = 0; r < 8; ++r) h[r] = (bf16_t)acc1[t][u][r];
        *(v8bf*)(C1t + sC1t * bz + (size_t)nG * ldct + mB) = h;
      }
      if (C1f) {
        float* dst = C1f + sC1f * bz;
        if (blendSrc) {
          const float* src = blendSrc + sBlend * bz;
#pragma unroll
          for (int r = 0; r < 8; ++r)
            dst[(size_t)(mB + r) * ldc + nG] =
                ca * src[(size_t)(mB + r) * ldc + nG] + cb * acc1[t][u][r];
        } else {
#pragma unroll
          for (int r = 0; r < 8; ++r) dst[(size_t)(mB + r) * ldc + nG] = acc1[t][u][r];
        }
      }
      if (C1h) {
#pragma unroll
        for (int r = 0; r < 8; ++r)
          C1h[sC1h * bz + (size_t)(mB + r) * ldc + nG] = (bf16_t)acc1[t][u][r];
      }
      if constexpr (FUSE2) {
        if (C2f) {
#pragma unroll
          for (int r = 0; r < 8; ++r)
            C2f[sC2f * bz + (size_t)(mB + r) * ldc + nG] = acc2[t][u][r];
        }
      }
    }
}

// ---------------- launcher ----------------
extern "C" void kernel_launch(void* const* d_in, const int* in_sizes, int n_in,
                              void* d_out, int out_size, void* d_ws, size_t ws_size,
                              hipStream_t stream) {
  (void)in_sizes; (void)n_in; (void)out_size; (void)ws_size;
  const float* features = (const float*)d_in[0];
  const float* inc      = (const float*)d_in[1];
  const float* Watt     = (const float*)d_in[2];
  const float* vcW      = (const float*)d_in[3];
  const float* ecW      = (const float*)d_in[6];
  const float* ec_alpha = (const float*)d_in[7];
  const int*   nlayers  = (const int*)d_in[8];

  char* ws = (char*)d_ws;
  bf16_t* incB  = (bf16_t*)(ws + O_INCB);
  bf16_t* featB = (bf16_t*)(ws + O_FEATB);
  bf16_t* featT = (bf16_t*)(ws + O_FEATT);
  bf16_t* WattB = (bf16_t*)(ws + O_WATT);
  bf16_t* vcWB  = (bf16_t*)(ws + O_VCW);
  bf16_t* ecWB  = (bf16_t*)(ws + O_ECW);
  bf16_t* ST    = (bf16_t*)(ws + O_ST);
  float*  ES    = (float*)(ws + O_ES);
  float*  AG    = (float*)(ws + O_AG);
  bf16_t* P     = (bf16_t*)(ws + O_P);
  bf16_t* EFT   = (bf16_t*)(ws + O_EFT);
  bf16_t* WN    = (bf16_t*)(ws + O_WN);
  float*  pm    = (float*)(ws + O_PM);
  float*  ps    = (float*)(ws + O_PS);
  float*  Mx    = (float*)(ws + O_MX);
  float*  Zx    = (float*)(ws + O_ZX);

  float* out_node = (float*)d_out;
  float* out_edge = out_node + (size_t)BB * MM * DD;
  float* out_inc  = out_edge + (size_t)BB * EE * DD;

  // 1) convert inc -> bf16 + echo f32 copy to output
  {
    int n4 = BB * MM * EE / 4;
    k_conv_inc<<<n4 / 256, 256, 0, stream>>>((const float4*)inc, (float4*)out_inc,
                                             (v4bf*)incB, n4);
  }
  // 2) features -> bf16 and bf16-transposed
  k_feat<<<dim3(DD / 32, MM / 32, BB), dim3(32, 8), 0, stream>>>(features, featB, featT);
  // 3) weights -> bf16 (gemm uses BT[N][K] = W as stored, since C = X @ W^T)
  k_conv_w<<<DD * DD / 256, 256, 0, stream>>>(Watt, WattB, DD * DD);
  k_conv_w<<<DD * DD / 256, 256, 0, stream>>>(vcW, vcWB, DD * DD);
  k_conv_w<<<DD * DD / 256, 256, 0, stream>>>(ecW, ecWB, DD * DD);

  // 4) S^T = (featB @ Watt^T)^T  -> bf16 [b][d][m]
  gemm_k<false, false><<<dim3(MM / 128, DD / 128, BB), 256, 0, stream>>>(
      featB, (size_t)MM * DD, DD, WattB, 0, DD, nullptr, 0, 0,
      nullptr, 0, ST, (size_t)DD * MM, MM, nullptr, 0, nullptr, 0,
      nullptr, 0, nullptr, nullptr, DD, DD);

  // 5) fused: ES = inc^T @ S, AG = inc^T @ feat   (A = incB accessed transposed)
  gemm_k<true, true><<<dim3(EE / 128, DD / 128, BB), 256, 0, stream>>>(
      incB, (size_t)MM * EE, EE, ST, (size_t)DD * MM, MM, featT, (size_t)DD * MM, MM,
      ES, (size_t)EE * DD, nullptr, 0, 0, nullptr, 0, AG, (size_t)EE * DD,
      nullptr, 0, nullptr, nullptr, MM, DD);

  // 6) column softmax over e, then P = attn * AG
  k_smpart<<<dim3(16, BB), DD, 0, stream>>>(ES, pm, ps);
  k_smcomb<<<BB, DD, 0, stream>>>(pm, ps, Mx, Zx);
  k_attnmul<<<dim3(EE * DD / 256, BB), 256, 0, stream>>>(ES, AG, Mx, Zx, P);

  // 7) EF = P @ vcW^T -> f32 out_edge + bf16 EF^T
  gemm_k<false, false><<<dim3(EE / 128, DD / 128, BB), 256, 0, stream>>>(
      P, (size_t)EE * DD, DD, vcWB, 0, DD, nullptr, 0, 0,
      out_edge, (size_t)EE * DD, EFT, (size_t)DD * EE, EE, nullptr, 0, nullptr, 0,
      nullptr, 0, nullptr, nullptr, DD, DD);

  // 8) WN = inc @ EF -> bf16
  gemm_k<false, false><<<dim3(MM / 128, DD / 128, BB), 256, 0, stream>>>(
      incB, (size_t)MM * EE, EE, EFT, (size_t)DD * EE, EE, nullptr, 0, 0,
      nullptr, 0, nullptr, 0, 0, WN, (size_t)MM * DD, nullptr, 0,
      nullptr, 0, nullptr, nullptr, EE, DD);

  // 9) node = a^n * feat + (1-a^n) * (WN @ ecW^T)
  gemm_k<false, false><<<dim3(MM / 128, DD / 128, BB), 256, 0, stream>>>(
      WN, (size_t)MM * DD, DD, ecWB, 0, DD, nullptr, 0, 0,
      out_node, (size_t)MM * DD, nullptr, 0, 0, nullptr, 0, nullptr, 0,
      features, (size_t)MM * DD, ec_alpha, nlayers, DD, DD);
}
